// GNNTransformer_47777216200788
// MI455X (gfx1250) — matrix operand order repaired and checked by reading, not compiled, not run on hardware
//
#include <hip/hip_runtime.h>
#include <hip/hip_bf16.h>
#include <math.h>

// ---------------- problem constants (from reference) ----------------
#define BB      4
#define NN      2048
#define TT      24
#define TPAD    32
#define FIN     16
#define HH      128
#define NHEAD   4
#define DH      32
#define EE      65536
#define EA      (EE + NN)        // edges + self loops = 67584
#define OUTT    12
#define BN      (BB * NN)        // 8192 sequences / nodes
#define HROWS   (BN * OUTT)      // 98304 head rows
#define SCALE_ATT 0.17677669529663688f  // 1/sqrt(32)

// ---------------- WMMA fragment types ----------------
typedef _Float16 v16h __attribute__((ext_vector_type(16)));
typedef _Float16 v8h  __attribute__((ext_vector_type(8)));
typedef float    v8f  __attribute__((ext_vector_type(8)));

// ---------------- CDNA5 async global->LDS copy (ASYNCcnt path) ----------------
// GLOBAL_LOAD_ASYNC_TO_LDS_B128, GV mode (saddr=off): LDS address taken from the VDST vgpr,
// 16 bytes per lane. LDS offset = low 32 bits of the flat pointer (ISA: LDS aperture
// address is truncated to addr[31:0]).
__device__ __forceinline__ void async_copy_b128(void* lds_dst, const void* gsrc) {
    asm volatile("global_load_async_to_lds_b128 %0, %1, off"
                 :: "v"((unsigned)(unsigned long long)lds_dst), "v"(gsrc)
                 : "memory");
}
__device__ __forceinline__ void async_wait_all() {
    asm volatile("s_wait_asynccnt 0" ::: "memory");
}

// A-fragment (16x32 f16) from an LDS tile, row stride `lda` halves.
// Layout: lanes 0-15 row M=lane, halves j -> K=j (0..15)
//         lanes 16-31 row M=lane-16, halves j -> K=16+j (16..31)
__device__ __forceinline__ v16h load_a_lds(const _Float16* A, int lda, int mt, int kt, int lane) {
    int row  = mt * 16 + (lane & 15);
    int koff = kt * 32 + ((lane & 16) ? 16 : 0);
    const _Float16* p = A + row * lda + koff;
    v8h lo = *(const v8h*)(p);
    v8h hi = *(const v8h*)(p + 8);
    v16h r;
#pragma unroll
    for (int i = 0; i < 8; ++i) { r[i] = lo[i]; r[i + 8] = hi[i]; }
    return r;
}

// B pre-packed: tile (nt,kt) stored contiguously, 512 halves, addr = ((nt*KT+kt)*512 + lane*16 + j)
// lane<16 holds K=0..15 of column nt*16+lane, lane>=16 holds K=16..31 of column nt*16+lane-16.
__device__ __forceinline__ v16h load_b_packed(const _Float16* Bp, int nt, int kt, int KT, int lane) {
    return *(const v16h*)(Bp + ((size_t)(nt * KT + kt)) * 512 + lane * 16);
}

__device__ __forceinline__ v8f gemm_tile(const _Float16* A, int lda, const _Float16* Bp,
                                         int mt, int nt, int KT, int lane) {
    v8f acc = {};
#pragma unroll 4
    for (int kt = 0; kt < KT; ++kt) {
        if (kt + 1 < KT)  // pull next packed B tile into near caches (global_prefetch_b8)
            __builtin_prefetch((const void*)(Bp + ((size_t)(nt * KT + kt + 1)) * 512 + lane * 16), 0, 3);
        v16h a = load_a_lds(A, lda, mt, kt, lane);
        v16h b = load_b_packed(Bp, nt, kt, KT, lane);
        acc = __builtin_amdgcn_wmma_f32_16x16x32_f16(false, a, false, b, (short)0, acc, false, false);
    }
    return acc;
}

// D tile (16x16 f32): lane<16 -> rows mt*16+0..7, col nt*16+lane; lane>=16 -> rows +8..15, col lane-16
__device__ __forceinline__ void store_tile_f16(v8f acc, _Float16* C, int ldc, int mt, int nt,
                                               int lane, const float* bias, bool relu) {
    int col = nt * 16 + (lane & 15);
    int rb  = mt * 16 + ((lane & 16) ? 8 : 0);
    float bv = bias ? bias[col] : 0.0f;
#pragma unroll
    for (int i = 0; i < 8; ++i) {
        float v = acc[i] + bv;
        if (relu) v = fmaxf(v, 0.0f);
        C[(rb + i) * ldc + col] = (_Float16)v;
    }
}

__device__ __forceinline__ void store_tile_res_f32(v8f acc, float* C, const _Float16* R, int ldc,
                                                   int mt, int nt, int lane, const float* bias) {
    int col = nt * 16 + (lane & 15);
    int rb  = mt * 16 + ((lane & 16) ? 8 : 0);
    float bv = bias[col];
#pragma unroll
    for (int i = 0; i < 8; ++i)
        C[(rb + i) * ldc + col] = acc[i] + bv + (float)R[(rb + i) * ldc + col];
}

__device__ __forceinline__ void ln_row_to_f16(const float* src, _Float16* dst,
                                              const float* g, const float* b, int n) {
    float m = 0.f;
    for (int c = 0; c < n; ++c) m += src[c];
    m /= (float)n;
    float v = 0.f;
    for (int c = 0; c < n; ++c) { float d = src[c] - m; v += d * d; }
    v /= (float)n;
    float inv = rsqrtf(v + 1e-5f);
    for (int c = 0; c < n; ++c) dst[c] = (_Float16)((src[c] - m) * inv * g[c] + b[c]);
}

// ---------------- weight packing: fp32 [K,N] -> f16 WMMA-B tiles ----------------
__global__ void k_pack(const float* __restrict__ src, _Float16* __restrict__ dst, int K, int N) {
    int total = K * N;
    int KT = K >> 5;
    for (int idx = blockIdx.x * blockDim.x + threadIdx.x; idx < total; idx += gridDim.x * blockDim.x) {
        int tile = idx >> 9, within = idx & 511;
        int lane = within >> 4, j = within & 15;
        int nt = tile / KT, kt = tile - nt * KT;
        int k = kt * 32 + ((lane & 16) ? 16 : 0) + j;
        int n = nt * 16 + (lane & 15);
        dst[idx] = (_Float16)src[(size_t)k * N + n];
    }
}

// ---------------- input projection: x[seq,24,16] @ W[16,128] + b -> h f16 [seq,32,128] ----------------
__global__ __launch_bounds__(256) void k_input_proj(const float* __restrict__ x,
                                                    const float* __restrict__ w,
                                                    const float* __restrict__ b,
                                                    _Float16* __restrict__ h) {
    __shared__ float sx[TT * FIN];
    __shared__ float sw[FIN * HH];
    int tid = threadIdx.x, seq = blockIdx.x;
    const float* xs = x + (size_t)seq * TT * FIN;
    for (int i = tid; i < TT * FIN; i += 256) sx[i] = xs[i];
    for (int i = tid; i < FIN * HH; i += 256) sw[i] = w[i];
    __syncthreads();
    _Float16* hs = h + (size_t)seq * TPAD * HH;
    for (int i = tid; i < TPAD * HH; i += 256) {
        int r = i >> 7, c = i & 127;
        float acc = 0.f;
        if (r < TT) {
            acc = b[c];
#pragma unroll
            for (int k = 0; k < FIN; ++k) acc += sx[r * FIN + k] * sw[k * HH + c];
        }
        hs[i] = (_Float16)acc;  // rows 24..31 zero-padded
    }
}

// ---------------- fused encoder layer: one workgroup (8 waves) per sequence ----------------
__global__ __launch_bounds__(256) void k_encoder_layer(
    _Float16* __restrict__ hbuf,
    const _Float16* __restrict__ wqkv_p, const _Float16* __restrict__ wo_p,
    const _Float16* __restrict__ w1_p,   const _Float16* __restrict__ w2_p,
    const float* __restrict__ bqkv, const float* __restrict__ bo,
    const float* __restrict__ ln1g, const float* __restrict__ ln1b,
    const float* __restrict__ b1,   const float* __restrict__ b2,
    const float* __restrict__ ln2g, const float* __restrict__ ln2b) {
    __shared__ _Float16 sh_h[TPAD * HH];          // 8 KB  activations (padded rows zero)
    __shared__ _Float16 sh_qkv[TPAD * 3 * HH];    // 24 KB qkv; later aliased as f32 scratch
    __shared__ _Float16 sh_o[TPAD * HH];          // 8 KB  attention output
    __shared__ _Float16 sh_mid[TPAD * 4 * HH];    // 32 KB FFN hidden
    float* sh_scr = (float*)sh_qkv;               // 16 KB f32 alias (qkv dead by then)

    int tid = threadIdx.x, lane = tid & 31, wv = tid >> 5;
    _Float16* hg = hbuf + (size_t)blockIdx.x * TPAD * HH;

    // load residual stream via async DMA: 4096 halves = 512 x b128, 2 per thread
    async_copy_b128(&sh_h[tid * 16],     hg + tid * 16);
    async_copy_b128(&sh_h[tid * 16 + 8], hg + tid * 16 + 8);
    async_wait_all();
    __syncthreads();

    // ---- QKV: [32,128] @ [128,384] -> sh_qkv; 48 tiles, 6/wave ----
    for (int t = wv * 6; t < wv * 6 + 6; ++t) {
        int mt = t & 1, nt = t >> 1;
        v8f acc = gemm_tile(sh_h, HH, wqkv_p, mt, nt, 4, lane);
        store_tile_f16(acc, sh_qkv, 3 * HH, mt, nt, lane, bqkv, false);
    }
    __syncthreads();

    // ---- attention (VALU fp32): 96 (head,row) tasks ----
    if (tid < NHEAD * TT) {
        int hd = tid / TT, r = tid % TT;
        const _Float16* qrow = sh_qkv + r * 384 + hd * DH;
        float s[TT];
        float mx = -1e30f;
#pragma unroll
        for (int j = 0; j < TT; ++j) {
            const _Float16* krow = sh_qkv + j * 384 + HH + hd * DH;
            float d = 0.f;
#pragma unroll
            for (int k = 0; k < DH; ++k) d += (float)qrow[k] * (float)krow[k];
            d *= SCALE_ATT;
            s[j] = d; mx = fmaxf(mx, d);
        }
        float sum = 0.f;
#pragma unroll
        for (int j = 0; j < TT; ++j) { s[j] = __expf(s[j] - mx); sum += s[j]; }
        float inv = 1.f / sum;
#pragma unroll 4
        for (int c = 0; c < DH; ++c) {
            float o = 0.f;
#pragma unroll
            for (int j = 0; j < TT; ++j) o += s[j] * (float)sh_qkv[j * 384 + 2 * HH + hd * DH + c];
            sh_o[r * HH + hd * DH + c] = (_Float16)(o * inv);
        }
    }
    if (tid >= 128) {  // zero pad rows 24..31 of sh_o
        int base = TT * HH + (tid - 128) * 8;
#pragma unroll
        for (int i = 0; i < 8; ++i) sh_o[base + i] = (_Float16)0.f;
    }
    __syncthreads();

    // ---- out-proj + residual -> f32 scratch; 16 tiles, 2/wave ----
    for (int t = wv * 2; t < wv * 2 + 2; ++t) {
        int mt = t & 1, nt = t >> 1;
        v8f acc = gemm_tile(sh_o, HH, wo_p, mt, nt, 4, lane);
        store_tile_res_f32(acc, sh_scr, sh_h, HH, mt, nt, lane, bo);
    }
    __syncthreads();
    if (tid < TT) ln_row_to_f16(sh_scr + tid * HH, sh_h + tid * HH, ln1g, ln1b, HH);
    __syncthreads();

    // ---- FFN1: [32,128]@[128,512], relu; 64 tiles, 8/wave ----
    for (int t = wv * 8; t < wv * 8 + 8; ++t) {
        int mt = t & 1, nt = t >> 1;
        v8f acc = gemm_tile(sh_h, HH, w1_p, mt, nt, 4, lane);
        store_tile_f16(acc, sh_mid, 4 * HH, mt, nt, lane, b1, true);
    }
    __syncthreads();

    // ---- FFN2: [32,512]@[512,128] + residual; 16 tiles, 2/wave, KT=16 ----
    for (int t = wv * 2; t < wv * 2 + 2; ++t) {
        int mt = t & 1, nt = t >> 1;
        v8f acc = gemm_tile(sh_mid, 4 * HH, w2_p, mt, nt, 16, lane);
        store_tile_res_f32(acc, sh_scr, sh_h, HH, mt, nt, lane, b2);
    }
    __syncthreads();
    if (tid < TT) ln_row_to_f16(sh_scr + tid * HH, hg + tid * HH, ln2g, ln2b, HH);
}

// ---------------- mean pool over T ----------------
__global__ void k_mean_pool(const _Float16* __restrict__ h, float* __restrict__ pooled) {
    int seq = blockIdx.x, c = threadIdx.x;  // 128 threads
    const _Float16* hs = h + (size_t)seq * TPAD * HH;
    float s = 0.f;
#pragma unroll
    for (int r = 0; r < TT; ++r) s += (float)hs[r * HH + c];
    pooled[(size_t)seq * HH + c] = s * (1.0f / TT);
}

// ---------------- GCN helpers ----------------
__global__ void k_zero(float* p, int n) {
    int i = blockIdx.x * blockDim.x + threadIdx.x;
    if (i < n) p[i] = 0.f;
}
__global__ void k_deg(const int* __restrict__ ei, float* __restrict__ deg) {
    int i = blockIdx.x * blockDim.x + threadIdx.x;
    if (i < EA) {
        int dst = (i < EE) ? ei[EE + i] : (i - EE);
        atomicAdd(&deg[dst], 1.0f);
    }
}
__global__ void k_dis(const float* __restrict__ deg, float* __restrict__ dis) {
    int i = blockIdx.x * blockDim.x + threadIdx.x;
    if (i < NN) dis[i] = rsqrtf(fmaxf(deg[i], 1.0f));
}
__global__ void k_gcn_transform(const float* __restrict__ in, const float* __restrict__ w,
                                float* __restrict__ out) {
    __shared__ float sx[HH];
    int row = blockIdx.x, c = threadIdx.x;  // 128 threads
    sx[c] = in[(size_t)row * HH + c];
    __syncthreads();
    float acc = 0.f;
#pragma unroll 8
    for (int k = 0; k < HH; ++k) acc += sx[k] * w[k * HH + c];
    out[(size_t)row * HH + c] = acc;
}
__global__ void k_gcn_scatter(const float* __restrict__ t, const int* __restrict__ ei,
                              const float* __restrict__ dis, float* __restrict__ out) {
    int e = blockIdx.x % EA, b = blockIdx.x / EA;
    int src, dst;
    if (e < EE) { src = ei[e]; dst = ei[EE + e]; } else { src = dst = e - EE; }
    float nrm = dis[src] * dis[dst];
    int c = threadIdx.x;  // 128 threads
    atomicAdd(&out[((size_t)b * NN + dst) * HH + c],
              t[((size_t)b * NN + src) * HH + c] * nrm);
}
__global__ void k_bias_relu(float* __restrict__ x, const float* __restrict__ b, int n) {
    int i = blockIdx.x * blockDim.x + threadIdx.x;
    if (i < n) x[i] = fmaxf(x[i] + b[i & (HH - 1)], 0.f);
}

// ---------------- build z = [enc || dec] f16 [98304, 256] ----------------
__global__ __launch_bounds__(256) void k_build_z(const float* __restrict__ enc,
                                                 const float* __restrict__ ftf,
                                                 const float* __restrict__ decw,
                                                 const float* __restrict__ decb,
                                                 _Float16* __restrict__ z) {
    __shared__ float sf[OUTT * 6];
    __shared__ float sw[6 * HH];
    __shared__ float sb[HH];
    int node = blockIdx.x, tid = threadIdx.x;
    const float* f = ftf + (size_t)node * OUTT * 6;
    if (tid < OUTT * 6) sf[tid] = f[tid];
    for (int i = tid; i < 6 * HH; i += 256) sw[i] = decw[i];
    if (tid < HH) sb[tid] = decb[tid];
    __syncthreads();
    _Float16* zr = z + (size_t)node * OUTT * 2 * HH;
    if (tid < HH) {
        _Float16 eh = (_Float16)enc[(size_t)node * HH + tid];
#pragma unroll
        for (int t = 0; t < OUTT; ++t) zr[t * 256 + tid] = eh;
    } else {
        int c = tid - HH;
#pragma unroll
        for (int t = 0; t < OUTT; ++t) {
            float s = sb[c];
#pragma unroll
            for (int k = 0; k < 6; ++k) s += sf[t * 6 + k] * sw[k * HH + c];
            zr[t * 256 + HH + c] = (_Float16)s;
        }
    }
}

// ---------------- fused head: WMMA o1 GEMM + LN + MLP tail ----------------
__global__ __launch_bounds__(256) void k_head(
    const _Float16* __restrict__ z, const _Float16* __restrict__ o1w_p,
    const float* __restrict__ o1b, const float* __restrict__ ln1g, const float* __restrict__ ln1b,
    const float* __restrict__ o2w, const float* __restrict__ o2b,
    const float* __restrict__ ln2g, const float* __restrict__ ln2b,
    const float* __restrict__ o3w, const float* __restrict__ o3b,
    const float* __restrict__ o4w, const float* __restrict__ o4b,
    float* __restrict__ out) {
    __shared__ _Float16 sh_a[32 * 256];   // 16 KB
    __shared__ float    sh_c[32 * HH];    // 16 KB
    __shared__ float    sh_w2[HH * 64];   // 32 KB
    __shared__ float    sh_w3[64 * 32];   // 8 KB
    __shared__ float    sh_w4[32];
    int tid = threadIdx.x, lane = tid & 31, wv = tid >> 5;
    size_t rowbase = (size_t)blockIdx.x * 32;

    // A-tile (32x256 f16 = 16 KB) via async DMA: 1024 x b128, 4 per thread
    const _Float16* zsrc = z + rowbase * 256;
#pragma unroll
    for (int c = 0; c < 4; ++c)
        async_copy_b128(&sh_a[(tid * 4 + c) * 8], zsrc + (tid * 4 + c) * 8);
    for (int i = tid; i < HH * 64; i += 256) sh_w2[i] = o2w[i];
    for (int i = tid; i < 64 * 32; i += 256) sh_w3[i] = o3w[i];
    if (tid < 32) sh_w4[tid] = o4w[tid];
    async_wait_all();
    __syncthreads();

    // o1: [32,256] @ [256,128]; 16 tiles, 2/wave, KT=8
    for (int t = wv * 2; t < wv * 2 + 2; ++t) {
        int mt = t & 1, nt = t >> 1;
        v8f acc = gemm_tile(sh_a, 256, o1w_p, mt, nt, 8, lane);
        int col = nt * 16 + (lane & 15);
        int rb  = mt * 16 + ((lane & 16) ? 8 : 0);
        float bv = o1b[col];
#pragma unroll
        for (int i = 0; i < 8; ++i) sh_c[(rb + i) * HH + col] = acc[i] + bv;
    }
    __syncthreads();

    if (tid < 32) {
        float* row = sh_c + tid * HH;
        // LN1 + relu (in place)
        float m = 0.f;
        for (int c = 0; c < HH; ++c) m += row[c];
        m *= (1.0f / HH);
        float v = 0.f;
        for (int c = 0; c < HH; ++c) { float d = row[c] - m; v += d * d; }
        v *= (1.0f / HH);
        float inv = rsqrtf(v + 1e-5f);
        for (int c = 0; c < HH; ++c) row[c] = fmaxf((row[c] - m) * inv * ln1g[c] + ln1b[c], 0.f);
        // o2 + LN2 + relu
        float a2[64];
        for (int j = 0; j < 64; ++j) {
            float s = o2b[j];
            for (int k = 0; k < HH; ++k) s += row[k] * sh_w2[k * 64 + j];
            a2[j] = s;
        }
        m = 0.f;
        for (int j = 0; j < 64; ++j) m += a2[j];
        m *= (1.0f / 64);
        v = 0.f;
        for (int j = 0; j < 64; ++j) { float d = a2[j] - m; v += d * d; }
        v *= (1.0f / 64);
        inv = rsqrtf(v + 1e-5f);
        for (int j = 0; j < 64; ++j) a2[j] = fmaxf((a2[j] - m) * inv * ln2g[j] + ln2b[j], 0.f);
        // o3 relu, o4
        float a3[32];
        for (int j = 0; j < 32; ++j) {
            float s = o3b[j];
            for (int k = 0; k < 64; ++k) s += a2[k] * sh_w3[k * 32 + j];
            a3[j] = fmaxf(s, 0.f);
        }
        float o = o4b[0];
        for (int k = 0; k < 32; ++k) o += a3[k] * sh_w4[k];
        out[rowbase + tid] = o;
    }
}

// ---------------- launcher ----------------
extern "C" void kernel_launch(void* const* d_in, const int* in_sizes, int n_in,
                              void* d_out, int out_size, void* d_ws, size_t ws_size,
                              hipStream_t stream) {
    (void)in_sizes; (void)n_in; (void)out_size; (void)ws_size;
    auto F = [&](int i) { return (const float*)d_in[i]; };

    const float* x   = F(0);
    const int*   ei  = (const int*)d_in[1];
    const float* ftf = F(2);
    int idx = 3;
    const float* inp_w = F(idx++); const float* inp_b = F(idx++);
    const float* L[2][12];
    for (int l = 0; l < 2; ++l)
        for (int j = 0; j < 12; ++j) L[l][j] = F(idx++);
    // per layer: 0 wqkv 1 bqkv 2 wo 3 bo 4 ln1g 5 ln1b 6 w1 7 b1 8 w2 9 b2 10 ln2g 11 ln2b
    const float* gcn1w = F(idx++); const float* gcn1b = F(idx++);
    const float* gcn2w = F(idx++); const float* gcn2b = F(idx++);
    const float* decw  = F(idx++); const float* decb  = F(idx++);
    const float* o1w   = F(idx++); const float* o1b   = F(idx++);
    const float* oln1g = F(idx++); const float* oln1b = F(idx++);
    const float* o2w   = F(idx++); const float* o2b   = F(idx++);
    const float* oln2g = F(idx++); const float* oln2b = F(idx++);
    const float* o3w   = F(idx++); const float* o3b   = F(idx++);
    const float* o4w   = F(idx++); const float* o4b   = F(idx++);

    // ---- workspace layout ----
    char* ws = (char*)d_ws;
    _Float16* hbuf  = (_Float16*)ws;                       // 64 MB: h f16 [8192,32,128]; reused for z
    _Float16* wpack = (_Float16*)(ws + (64ull << 20));     // 1 MB packed weights
    float* pooled   = (float*)(ws + (65ull << 20));        // 4 MB
    float* gt       = (float*)(ws + (69ull << 20));        // 4 MB gcn transform tmp
    float* g1       = (float*)(ws + (73ull << 20));        // 4 MB
    float* g2       = (float*)(ws + (77ull << 20));        // 4 MB
    float* deg      = (float*)(ws + (81ull << 20));
    float* dis      = deg + 4096;

    _Float16* wqkv_p[2], *wo_p[2], *w1_p[2], *w2_p[2];
    for (int l = 0; l < 2; ++l) {
        _Float16* base = wpack + (size_t)l * 196608;
        wqkv_p[l] = base;            // 128x384 = 49152
        wo_p[l]   = base + 49152;    // 128x128 = 16384
        w1_p[l]   = base + 65536;    // 128x512 = 65536
        w2_p[l]   = base + 131072;   // 512x128 = 65536
    }
    _Float16* o1_p = wpack + 393216; // 256x128 = 32768

    // ---- pack all GEMM weights into WMMA-B f16 layout ----
    auto pack = [&](const float* s, _Float16* d, int K, int N) {
        int tot = K * N;
        k_pack<<<(tot + 255) / 256, 256, 0, stream>>>(s, d, K, N);
    };
    for (int l = 0; l < 2; ++l) {
        pack(L[l][0], wqkv_p[l], 128, 384);
        pack(L[l][2], wo_p[l],   128, 128);
        pack(L[l][6], w1_p[l],   128, 512);
        pack(L[l][8], w2_p[l],   512, 128);
    }
    pack(o1w, o1_p, 256, 128);

    // ---- encoder ----
    k_input_proj<<<BN, 256, 0, stream>>>(x, inp_w, inp_b, hbuf);
    for (int l = 0; l < 2; ++l)
        k_encoder_layer<<<BN, 256, 0, stream>>>(hbuf,
            wqkv_p[l], wo_p[l], w1_p[l], w2_p[l],
            L[l][1], L[l][3], L[l][4], L[l][5], L[l][7], L[l][9], L[l][10], L[l][11]);
    k_mean_pool<<<BN, 128, 0, stream>>>(hbuf, pooled);

    // ---- GCN normalization ----
    k_zero<<<(NN + 255) / 256, 256, 0, stream>>>(deg, NN);
    k_deg<<<(EA + 255) / 256, 256, 0, stream>>>(ei, deg);
    k_dis<<<(NN + 255) / 256, 256, 0, stream>>>(deg, dis);

    const int HN = BN * HH;  // 1048576
    // GCN layer 1
    k_gcn_transform<<<BN, HH, 0, stream>>>(pooled, gcn1w, gt);
    k_zero<<<(HN + 255) / 256, 256, 0, stream>>>(g1, HN);
    k_gcn_scatter<<<BB * EA, HH, 0, stream>>>(gt, ei, dis, g1);
    k_bias_relu<<<(HN + 255) / 256, 256, 0, stream>>>(g1, gcn1b, HN);
    // GCN layer 2
    k_gcn_transform<<<BN, HH, 0, stream>>>(g1, gcn2w, gt);
    k_zero<<<(HN + 255) / 256, 256, 0, stream>>>(g2, HN);
    k_gcn_scatter<<<BB * EA, HH, 0, stream>>>(gt, ei, dis, g2);
    k_bias_relu<<<(HN + 255) / 256, 256, 0, stream>>>(g2, gcn2b, HN);

    // ---- head (z reuses the h region; encoder is finished) ----
    _Float16* z = hbuf;
    k_build_z<<<BN, 256, 0, stream>>>(g2, ftf, decw, decb, z);
    k_head<<<HROWS / 32, 256, 0, stream>>>(z, o1_p, o1b, oln1g, oln1b,
                                           o2w, o2b, oln2g, oln2b,
                                           o3w, o3b, o4w, o4b, (float*)d_out);
}
